// GraphAttentionLayer_87196426043787
// MI455X (gfx1250) — compile-verified, matrix operand order checked by reading, not run on hardware
//
#include <hip/hip_runtime.h>
#include <hip/hip_bf16.h>

typedef __attribute__((ext_vector_type(16))) _Float16 v16h;
typedef __attribute__((ext_vector_type(8)))  float    v8f;

#define IN_DIM   128
#define FEATS    128   // HEADS*OUT_DIM
#define HEADS    4
#define CPH      32    // channels per head
#define NEG_SLOPE 0.2f
#define LN_EPS    1e-5f
#define SM_EPS    1e-16f

// ---------------------------------------------------------------------------
// K0: zero the output accumulator, init segmax keys (monotonic-uint -inf = 0)
// and softmax denominators.  Must run every launch (harness doesn't re-poison).
// ---------------------------------------------------------------------------
__global__ void gat_init(float* __restrict__ out, unsigned* __restrict__ segmax,
                         float* __restrict__ denom, int N) {
  int i = blockIdx.x * 256 + threadIdx.x;
  if (i < N * FEATS) out[i] = 0.0f;
  if (i < N * HEADS) { segmax[i] = 0u; denom[i] = 0.0f; }
}

// ---------------------------------------------------------------------------
// K1: pack W[k][n] (row-major 128x128 f32) into the CDNA5 16-bit B-operand
// lane layout, per (ntile, kchunk, lane): lane<16 holds K[0..7]++K[16..23]
// of column n = t*16+(lane&15); lane>=16 holds K[8..15]++K[24..31].
// ---------------------------------------------------------------------------
__global__ void gat_pack_w(const float* __restrict__ W, _Float16* __restrict__ Wp) {
  int idx = blockIdx.x * 256 + threadIdx.x;            // 8 tiles * 4 chunks * 32 lanes
  if (idx >= 8 * 4 * 32) return;
  int lane = idx & 31;
  int q    = (idx >> 5) & 3;
  int t    = idx >> 7;
  int n     = t * 16 + (lane & 15);
  int kbase = q * 32 + (lane >> 4) * 8;
  _Float16* o = Wp + idx * 16;
#pragma unroll
  for (int j = 0; j < 8; ++j) {
    o[j]     = (_Float16)W[(kbase + j) * FEATS + n];
    o[j + 8] = (_Float16)W[(kbase + 16 + j) * FEATS + n];
  }
}

// ---------------------------------------------------------------------------
// K2: h = x @ W with v_wmma_f32_16x16x32_f16.  Block = 256 threads = 8 waves;
// wave w owns the 16x16 tile at column tile w; block owns 16 rows.
// Grid = N/16 (N = 100000 is a multiple of 16, no bounds checks, EXEC all 1s).
// ---------------------------------------------------------------------------
__global__ void gat_gemm(const float* __restrict__ x, const _Float16* __restrict__ Wp,
                         float* __restrict__ h) {
  const int wave   = threadIdx.x >> 5;     // column tile 0..7
  const int lane   = threadIdx.x & 31;
  const int m      = lane & 15;
  const int hiHalf = lane >> 4;            // 0 or 1
  const int rowb   = blockIdx.x * 16;
  const float* xrow = x + (size_t)(rowb + m) * IN_DIM;

  v8f acc = {};
#pragma unroll
  for (int q = 0; q < 4; ++q) {
    const int kbase = q * 32 + hiHalf * 8;
    v16h a;
#pragma unroll
    for (int j = 0; j < 8; ++j) {
      a[j]     = (_Float16)xrow[kbase + j];
      a[j + 8] = (_Float16)xrow[kbase + 16 + j];
    }
    v16h b = *(const v16h*)(Wp + (((wave * 4) + q) * 32 + lane) * 16);
    acc = __builtin_amdgcn_wmma_f32_16x16x32_f16(
        /*neg_a=*/false, a, /*neg_b=*/false, b,
        /*c_mod=*/(short)0, acc, /*reuse_a=*/false, /*reuse_b=*/false);
  }
  const int col = wave * 16 + m;
#pragma unroll
  for (int r = 0; r < 8; ++r)
    h[(size_t)(rowb + r + hiHalf * 8) * FEATS + col] = acc[r];
}

// ---------------------------------------------------------------------------
// K3: per (node, head) attention scores a_src/a_dst = <h_row_head, att_vec>.
// ---------------------------------------------------------------------------
__global__ void gat_scores(const float* __restrict__ h, const float* __restrict__ att_src,
                           const float* __restrict__ att_dst,
                           float* __restrict__ a_src, float* __restrict__ a_dst, int NH) {
  int idx = blockIdx.x * 256 + threadIdx.x;
  if (idx >= NH) return;
  int hd = idx & (HEADS - 1);
  int n  = idx >> 2;
  const float* row = h + (size_t)n * FEATS + hd * CPH;
  const float* as  = att_src + hd * CPH;
  const float* ad  = att_dst + hd * CPH;
  float s = 0.0f, t = 0.0f;
#pragma unroll
  for (int j = 0; j < 8; ++j) {
    float4 hv = *(const float4*)(row + j * 4);
    float4 a4 = *(const float4*)(as + j * 4);
    float4 b4 = *(const float4*)(ad + j * 4);
    s += hv.x * a4.x + hv.y * a4.y + hv.z * a4.z + hv.w * a4.w;
    t += hv.x * b4.x + hv.y * b4.y + hv.z * b4.z + hv.w * b4.w;
  }
  a_src[idx] = s;
  a_dst[idx] = t;
}

__device__ __forceinline__ float leaky(float v) { return v > 0.0f ? v : NEG_SLOPE * v; }
// order-preserving float<->uint so integer atomicMax implements float max
__device__ __forceinline__ unsigned fkey(float f) {
  unsigned b = __float_as_uint(f);
  return (b & 0x80000000u) ? ~b : (b | 0x80000000u);
}

// ---------------------------------------------------------------------------
// K4: per-edge leaky-relu logits -> segment max via integer atomicMax.
// Self loops are edges [E, Etot).
// ---------------------------------------------------------------------------
__global__ void gat_edge_max(const int* __restrict__ srcs, const int* __restrict__ dsts,
                             int E, int Etot, const float* __restrict__ a_src,
                             const float* __restrict__ a_dst, unsigned* __restrict__ segmax) {
  int i = blockIdx.x * 256 + threadIdx.x;
  if (i >= Etot) return;
  int s, d;
  if (i < E) { s = srcs[i]; d = dsts[i]; } else { s = d = i - E; }
  float4 ls = *(const float4*)(a_src + (size_t)s * HEADS);
  float4 ld = *(const float4*)(a_dst + (size_t)d * HEADS);
  unsigned* p = segmax + (size_t)d * HEADS;
  atomicMax(p + 0, fkey(leaky(ls.x + ld.x)));
  atomicMax(p + 1, fkey(leaky(ls.y + ld.y)));
  atomicMax(p + 2, fkey(leaky(ls.z + ld.z)));
  atomicMax(p + 3, fkey(leaky(ls.w + ld.w)));
}

// K4b: decode monotonic keys back to floats, in place.
__global__ void gat_decode_max(unsigned* __restrict__ segmax, int NH) {
  int i = blockIdx.x * 256 + threadIdx.x;
  if (i >= NH) return;
  unsigned k = segmax[i];
  unsigned b = (k & 0x80000000u) ? (k ^ 0x80000000u) : ~k;
  ((float*)segmax)[i] = __uint_as_float(b);
}

// ---------------------------------------------------------------------------
// K5: e = exp(logit - segmax[dst]); denom[dst] += e  (float atomicAdd).
// ---------------------------------------------------------------------------
__global__ void gat_edge_expsum(const int* __restrict__ srcs, const int* __restrict__ dsts,
                                int E, int Etot, const float* __restrict__ a_src,
                                const float* __restrict__ a_dst,
                                const float* __restrict__ segmax, float* __restrict__ denom) {
  int i = blockIdx.x * 256 + threadIdx.x;
  if (i >= Etot) return;
  int s, d;
  if (i < E) { s = srcs[i]; d = dsts[i]; } else { s = d = i - E; }
  float4 ls = *(const float4*)(a_src + (size_t)s * HEADS);
  float4 ld = *(const float4*)(a_dst + (size_t)d * HEADS);
  float4 mx = *(const float4*)(segmax + (size_t)d * HEADS);
  float* p = denom + (size_t)d * HEADS;
  atomicAdd(p + 0, __expf(leaky(ls.x + ld.x) - mx.x));
  atomicAdd(p + 1, __expf(leaky(ls.y + ld.y) - mx.y));
  atomicAdd(p + 2, __expf(leaky(ls.z + ld.z) - mx.z));
  atomicAdd(p + 3, __expf(leaky(ls.w + ld.w) - mx.w));
}

// ---------------------------------------------------------------------------
// K6: one wave32 per edge: out[dst] += h[src] * alpha(head).  Lane owns 4
// consecutive channels (float4 gather of h[src], 4 f32 atomics into L2).
// ---------------------------------------------------------------------------
__global__ void gat_edge_scatter(const int* __restrict__ srcs, const int* __restrict__ dsts,
                                 int E, int Etot, const float* __restrict__ a_src,
                                 const float* __restrict__ a_dst,
                                 const float* __restrict__ segmax, const float* __restrict__ denom,
                                 const float* __restrict__ h, float* __restrict__ out) {
  int e = blockIdx.x * 8 + (threadIdx.x >> 5);
  if (e >= Etot) return;
  int lane = threadIdx.x & 31;
  int s, d;
  if (e < E) { s = srcs[e]; d = dsts[e]; } else { s = d = e - E; }
  int hd = lane >> 3;                         // 32 lanes * 4ch = 128 ch; 8 lanes/head
  float l = leaky(a_src[(size_t)s * HEADS + hd] + a_dst[(size_t)d * HEADS + hd]);
  float alpha = __expf(l - segmax[(size_t)d * HEADS + hd]) /
                (denom[(size_t)d * HEADS + hd] + SM_EPS);
  float4 hv = *(const float4*)(h + (size_t)s * FEATS + lane * 4);
  float* op = out + (size_t)d * FEATS + lane * 4;
  atomicAdd(op + 0, hv.x * alpha);
  atomicAdd(op + 1, hv.y * alpha);
  atomicAdd(op + 2, hv.z * alpha);
  atomicAdd(op + 3, hv.w * alpha);
}

// ---------------------------------------------------------------------------
// K7: out = relu(layernorm(out + bias) * gamma + beta).  One wave per node,
// wave32 shuffle reduction for mean/var.  In-place on d_out.
// ---------------------------------------------------------------------------
__global__ void gat_finalize(float* __restrict__ out, const float* __restrict__ bias,
                             const float* __restrict__ gamma, const float* __restrict__ beta,
                             int N) {
  int n = blockIdx.x * 8 + (threadIdx.x >> 5);
  if (n >= N) return;
  int lane = threadIdx.x & 31;
  float* row = out + (size_t)n * FEATS + lane * 4;
  float4 v = *(float4*)row;
  float4 b = *(const float4*)(bias + lane * 4);
  v.x += b.x; v.y += b.y; v.z += b.z; v.w += b.w;
  float s  = v.x + v.y + v.z + v.w;
  float ss = v.x * v.x + v.y * v.y + v.z * v.z + v.w * v.w;
#pragma unroll
  for (int off = 16; off > 0; off >>= 1) {
    s  += __shfl_xor(s, off, 32);
    ss += __shfl_xor(ss, off, 32);
  }
  float mu  = s * (1.0f / FEATS);
  float var = ss * (1.0f / FEATS) - mu * mu;
  float inv = rsqrtf(var + LN_EPS);
  float4 g  = *(const float4*)(gamma + lane * 4);
  float4 be = *(const float4*)(beta + lane * 4);
  float4 y;
  y.x = fmaxf((v.x - mu) * inv * g.x + be.x, 0.0f);
  y.y = fmaxf((v.y - mu) * inv * g.y + be.y, 0.0f);
  y.z = fmaxf((v.z - mu) * inv * g.z + be.z, 0.0f);
  y.w = fmaxf((v.w - mu) * inv * g.w + be.w, 0.0f);
  *(float4*)row = y;
}

extern "C" void kernel_launch(void* const* d_in, const int* in_sizes, int n_in,
                              void* d_out, int out_size, void* d_ws, size_t ws_size,
                              hipStream_t stream) {
  const float* x       = (const float*)d_in[0];
  const int*   ei      = (const int*)d_in[1];    // int32 (jax default x64-disabled)
  const float* W       = (const float*)d_in[2];
  const float* att_src = (const float*)d_in[3];
  const float* att_dst = (const float*)d_in[4];
  const float* bias    = (const float*)d_in[5];
  const float* gamma   = (const float*)d_in[6];
  const float* beta    = (const float*)d_in[7];
  float* out = (float*)d_out;

  const int N    = in_sizes[0] / IN_DIM;
  const int E    = in_sizes[1] / 2;
  const int Etot = E + N;                         // + self loops
  const int* srcs = ei;
  const int* dsts = ei + E;

  // workspace carve-up
  char* ws = (char*)d_ws;
  float*     h_buf  = (float*)ws;                       ws += (size_t)N * FEATS * 4;
  _Float16*  Wp     = (_Float16*)ws;                    ws += 8 * 4 * 32 * 16 * 2;
  float*     as_buf = (float*)ws;                       ws += (size_t)N * HEADS * 4;
  float*     ad_buf = (float*)ws;                       ws += (size_t)N * HEADS * 4;
  unsigned*  segmax = (unsigned*)ws;                    ws += (size_t)N * HEADS * 4;
  float*     denom  = (float*)ws;

  const int blk = 256;
  gat_init<<<(N * FEATS + blk - 1) / blk, blk, 0, stream>>>(out, segmax, denom, N);
  gat_pack_w<<<(8 * 4 * 32 + blk - 1) / blk, blk, 0, stream>>>(W, Wp);
  gat_gemm<<<N / 16, blk, 0, stream>>>(x, Wp, h_buf);
  gat_scores<<<(N * HEADS + blk - 1) / blk, blk, 0, stream>>>(h_buf, att_src, att_dst,
                                                              as_buf, ad_buf, N * HEADS);
  gat_edge_max<<<(Etot + blk - 1) / blk, blk, 0, stream>>>(srcs, dsts, E, Etot,
                                                           as_buf, ad_buf, segmax);
  gat_decode_max<<<(N * HEADS + blk - 1) / blk, blk, 0, stream>>>(segmax, N * HEADS);
  gat_edge_expsum<<<(Etot + blk - 1) / blk, blk, 0, stream>>>(srcs, dsts, E, Etot,
                                                              as_buf, ad_buf,
                                                              (const float*)segmax, denom);
  gat_edge_scatter<<<(Etot + 7) / 8, blk, 0, stream>>>(srcs, dsts, E, Etot,
                                                       as_buf, ad_buf,
                                                       (const float*)segmax, denom,
                                                       h_buf, out);
  gat_finalize<<<(N + 7) / 8, blk, 0, stream>>>(out, bias, gamma, beta, N);
}
